// PoplarCrossOmicsModel_88785563943721
// MI455X (gfx1250) — matrix operand order repaired and checked by reading, not compiled, not run on hardware
//
#include <hip/hip_runtime.h>
#include <hip/hip_bf16.h>
#include <stdint.h>

// ---------------------------------------------------------------------------
// Problem constants (from the reference)
// ---------------------------------------------------------------------------
#define N_NODES 200000
#define D 256
#define P_PAIRS 65536
#define KNEG 8
#define ROWS_PB 32   // node rows per workgroup in the projection GEMM

typedef __attribute__((ext_vector_type(16))) __bf16 v16bf;
typedef __attribute__((ext_vector_type(8)))  float  v8f;

__device__ __forceinline__ unsigned short f2bf(float f) {
    uint32_t u = __float_as_uint(f);
    uint32_t r = u + 0x7FFFu + ((u >> 16) & 1u);   // round-to-nearest-even
    return (unsigned short)(r >> 16);
}

// ---------------------------------------------------------------------------
// Kernel 0: convert W_src / W_dst (f32, row-major [j][k]) to bf16 in ws
// ---------------------------------------------------------------------------
__global__ void convert_w_kernel(const float* __restrict__ Ws,
                                 const float* __restrict__ Wd,
                                 unsigned short* __restrict__ Wsb,
                                 unsigned short* __restrict__ Wdb) {
    int i = blockIdx.x * 256 + threadIdx.x;
    Wsb[i] = f2bf(Ws[i]);
    Wdb[i] = f2bf(Wd[i]);
}

// ---------------------------------------------------------------------------
// Kernel 1: U = l2norm(l2norm(emb) @ Wsrc^T), V likewise, stored bf16.
// Block: 256 threads (8 waves), 32 node rows.
// Wave task: side = wave&1 (U/V), mt = (wave>>1)&1 (16-row tile),
//            nh = wave>>2 (128-col half). 8 WMMA n-tiles x 8 k-steps.
// W K-slices are double-buffered in LDS via GLOBAL_LOAD_ASYNC_TO_LDS_B128
// (ASYNCcnt-tracked, direct-to-LDS) so slice kt+1 streams in while the 64
// v_wmma_f32_16x16x32_bf16 ops of slice kt execute.
// ---------------------------------------------------------------------------
__global__ __launch_bounds__(256)
void proj_norm_kernel(const float* __restrict__ emb,
                      const unsigned short* __restrict__ Wsb,
                      const unsigned short* __restrict__ Wdb,
                      unsigned short* __restrict__ Ub,
                      unsigned short* __restrict__ Vb) {
    __shared__ unsigned short zsh[ROWS_PB][264];     // normalized rows, bf16, padded
    __shared__ unsigned short wsh[2][2][256][40];    // double-buffered K-slice, both W's
    __shared__ float rowSS[2][ROWS_PB];              // per-output-row sum of squares

    const int tid  = threadIdx.x;
    const int base = blockIdx.x * ROWS_PB;

    // async stage of one 32-wide K-slice of both weight matrices into LDS:
    // 2048 chunks of 8 bf16 (16B) -> 8 async b128 ops per thread (per wave).
    auto issue_slice = [&](int kt, int buf) {
#pragma unroll
        for (int it = 0; it < 8; ++it) {
            int idx = tid + it * 256;
            int sd  = idx >> 10;           // side (W_src / W_dst)
            int rem = idx & 1023;
            int j   = rem >> 2;            // output column (W row)
            int kk  = (rem & 3) << 3;      // 0,8,16,24
            const unsigned short* gp =
                (sd ? Wdb : Wsb) + (size_t)j * D + kt * 32 + kk;
            uint32_t lds = (uint32_t)(uintptr_t)&wsh[buf][sd][j][kk];
            asm volatile("global_load_async_to_lds_b128 %0, %1, off"
                         :: "v"(lds), "v"(gp) : "memory");
        }
    };

    // kick off slice 0 before doing any math: overlaps the normalize stage
    issue_slice(0, 0);

    // ---- stage 1: l2-normalize 32 embedding rows into LDS as bf16 ----
    {
        const int row = tid >> 3;          // 32 rows x 8 threads
        const int seg = tid & 7;           // 32 floats per thread
        const float* src = emb + (size_t)(base + row) * D + seg * 32;
        float v[32];
        float ss = 0.f;
#pragma unroll
        for (int i = 0; i < 32; ++i) { v[i] = src[i]; ss += v[i] * v[i]; }
#pragma unroll
        for (int m = 1; m < 8; m <<= 1) ss += __shfl_xor(ss, m, 8);
        const float inv = 1.0f / fmaxf(sqrtf(ss), 1e-12f);
#pragma unroll
        for (int i = 0; i < 32; ++i) zsh[row][seg * 32 + i] = f2bf(v[i] * inv);
    }
    if (tid < 64) ((float*)rowSS)[tid] = 0.f;

    const int wave = tid >> 5;
    const int lane = tid & 31;
    const int lr   = lane & 15;
    const int hi   = lane >> 4;
    const int side = wave & 1;
    const int mt   = (wave >> 1) & 1;
    const int nh   = wave >> 2;

    v8f acc[8];
#pragma unroll
    for (int nt = 0; nt < 8; ++nt) { v8f z = {}; acc[nt] = z; }

    // ---- stage 2: K loop, 32 per step, software-pipelined ----
    for (int kt = 0; kt < 8; ++kt) {
        const int buf = kt & 1;
        if (kt < 7) {
            issue_slice(kt + 1, buf ^ 1);
            // in-order completion: cnt<=8 means the oldest batch (slice kt) landed
            asm volatile("s_wait_asynccnt 0x8" ::: "memory");
        } else {
            asm volatile("s_wait_asynccnt 0x0" ::: "memory");
        }
        __syncthreads();   // publish slice kt LDS writes to all waves

        // A fragment: rows mt*16 + lr, VGPR g -> k = 2g+16*hi (contiguous, matches B)
        v16bf afrag;
        {
            uint32_t* ap = (uint32_t*)&afrag;
#pragma unroll
            for (int g = 0; g < 8; ++g) {
                int kb = 2 * g + 16 * hi;
                ap[g] = *(const uint32_t*)&zsh[mt * 16 + lr][kt * 32 + kb];
            }
        }
#pragma unroll
        for (int nt = 0; nt < 8; ++nt) {
            const int nb = nh * 128 + nt * 16;
            // B fragment: B[k][n] = W[nb+n][kt*32+k]; lane n = lr, VGPR g -> k = 2g+16*hi
            v16bf bfrag;
            uint32_t* bp = (uint32_t*)&bfrag;
#pragma unroll
            for (int g = 0; g < 8; ++g) {
                bp[g] = *(const uint32_t*)&wsh[buf][side][nb + lr][2 * g + 16 * hi];
            }
            acc[nt] = __builtin_amdgcn_wmma_f32_16x16x32_bf16(
                false, afrag, false, bfrag, (short)0, acc[nt], false, false);
        }
        __syncthreads();   // all waves done reading buf before it is re-issued
    }

    // ---- stage 3: per-row sum of squares (this wave covers 128 of 256 cols) ----
#pragma unroll
    for (int g = 0; g < 8; ++g) {
        float ssg = 0.f;
#pragma unroll
        for (int nt = 0; nt < 8; ++nt) { float x = acc[nt][g]; ssg += x * x; }
#pragma unroll
        for (int m = 1; m < 16; m <<= 1) ssg += __shfl_xor(ssg, m, 16);
        if (lr == 0) atomicAdd(&rowSS[side][mt * 16 + g + 8 * hi], ssg);
    }
    __syncthreads();

    // ---- stage 4: scale + store bf16 results ----
    unsigned short* outp = side ? Vb : Ub;
#pragma unroll
    for (int g = 0; g < 8; ++g) {
        const int r = mt * 16 + g + 8 * hi;                 // C/D layout: M = g + 8*hi
        const float inv = 1.0f / fmaxf(sqrtf(rowSS[side][r]), 1e-12f);
        const size_t rowoff = (size_t)(base + r) * D;
#pragma unroll
        for (int nt = 0; nt < 8; ++nt) {
            const int col = nh * 128 + nt * 16 + lr;        // C/D layout: N = lane&15
            outp[rowoff + col] = f2bf(acc[nt][g] * inv);
        }
    }
}

// ---------------------------------------------------------------------------
// Kernel 2: one wave per positive pair: 9 gathered dots -> masked CE loss row
// ---------------------------------------------------------------------------
__global__ __launch_bounds__(256)
void pair_loss_kernel(const unsigned short* __restrict__ Ub,
                      const unsigned short* __restrict__ Vb,
                      const int* __restrict__ pos_src, const int* __restrict__ pos_dst,
                      const int* __restrict__ pos_sign,
                      const int* __restrict__ neg_src, const int* __restrict__ neg_dst,
                      float* __restrict__ lossArr, float* __restrict__ maskArr) {
    const int tid  = threadIdx.x;
    const int lane = tid & 31;
    const int p    = blockIdx.x * 8 + (tid >> 5);

    float sims[9];
#pragma unroll
    for (int e = 0; e < 9; ++e) {
        int s, d;
        if (e == 0) { s = pos_src[p];                      d = pos_dst[p]; }
        else        { s = neg_src[(size_t)p * KNEG + e-1]; d = neg_dst[(size_t)p * KNEG + e-1]; }
        const uint4 ua = *(const uint4*)(Ub + (size_t)s * D + lane * 8);  // 8 bf16
        const uint4 va = *(const uint4*)(Vb + (size_t)d * D + lane * 8);
        const uint32_t* uw = (const uint32_t*)&ua;
        const uint32_t* vw = (const uint32_t*)&va;
        float dot = 0.f;
#pragma unroll
        for (int w = 0; w < 4; ++w) {
            float u0 = __uint_as_float(uw[w] << 16);
            float u1 = __uint_as_float(uw[w] & 0xFFFF0000u);
            float v0 = __uint_as_float(vw[w] << 16);
            float v1 = __uint_as_float(vw[w] & 0xFFFF0000u);
            dot = fmaf(u0, v0, dot);
            dot = fmaf(u1, v1, dot);
        }
#pragma unroll
        for (int m = 1; m < 32; m <<= 1) dot += __shfl_xor(dot, m, 32);
        sims[e] = dot;
    }

    const int sign = pos_sign[p];
    sims[0] *= (sign == 0) ? -1.0f : 1.0f;

    float l[9], mx = -1e30f;
#pragma unroll
    for (int e = 0; e < 9; ++e) { l[e] = sims[e] * 5.0f; mx = fmaxf(mx, l[e]); } // /TEMP
    float se = 0.f;
#pragma unroll
    for (int e = 0; e < 9; ++e) se += __expf(l[e] - mx);
    const float loss = mx + __logf(se) - l[0];
    const float msk  = (sign != 2) ? 1.0f : 0.0f;

    if (lane == 0) { lossArr[p] = loss * msk; maskArr[p] = msk; }
}

// ---------------------------------------------------------------------------
// Deterministic two-stage reduction -> scalar output
// ---------------------------------------------------------------------------
__global__ void reduce_stage1(const float* __restrict__ lossArr,
                              const float* __restrict__ maskArr,
                              float* __restrict__ pL, float* __restrict__ pM) {
    __shared__ float sL[256], sM[256];
    const int t = threadIdx.x;
    const int i = blockIdx.x * 256 + t;
    sL[t] = lossArr[i];
    sM[t] = maskArr[i];
    __syncthreads();
    for (int s = 128; s > 0; s >>= 1) {
        if (t < s) { sL[t] += sL[t + s]; sM[t] += sM[t + s]; }
        __syncthreads();
    }
    if (t == 0) { pL[blockIdx.x] = sL[0]; pM[blockIdx.x] = sM[0]; }
}

__global__ void reduce_stage2(const float* __restrict__ pL,
                              const float* __restrict__ pM,
                              float* __restrict__ out) {
    __shared__ float sL[256], sM[256];
    const int t = threadIdx.x;
    sL[t] = pL[t];
    sM[t] = pM[t];
    __syncthreads();
    for (int s = 128; s > 0; s >>= 1) {
        if (t < s) { sL[t] += sL[t + s]; sM[t] += sM[t + s]; }
        __syncthreads();
    }
    if (t == 0) out[0] = sL[0] / (sM[0] + 1e-9f);
}

// ---------------------------------------------------------------------------
extern "C" void kernel_launch(void* const* d_in, const int* in_sizes, int n_in,
                              void* d_out, int out_size, void* d_ws, size_t ws_size,
                              hipStream_t stream) {
    (void)in_sizes; (void)n_in; (void)out_size; (void)ws_size;

    const float* emb     = (const float*)d_in[0];
    const float* Ws      = (const float*)d_in[1];
    const float* Wd      = (const float*)d_in[2];
    const int* pos_src   = (const int*)d_in[3];
    const int* pos_dst   = (const int*)d_in[4];
    const int* pos_sign  = (const int*)d_in[5];
    const int* neg_src   = (const int*)d_in[6];
    const int* neg_dst   = (const int*)d_in[7];

    // workspace carve-up (all 16B aligned): ~205.6 MB total
    char* ws = (char*)d_ws;
    const size_t NB = (size_t)N_NODES * D * sizeof(unsigned short); // 102.4 MB
    unsigned short* Ub  = (unsigned short*)ws;  ws += NB;
    unsigned short* Vb  = (unsigned short*)ws;  ws += NB;
    unsigned short* Wsb = (unsigned short*)ws;  ws += (size_t)D * D * sizeof(unsigned short);
    unsigned short* Wdb = (unsigned short*)ws;  ws += (size_t)D * D * sizeof(unsigned short);
    float* lossArr = (float*)ws;  ws += (size_t)P_PAIRS * sizeof(float);
    float* maskArr = (float*)ws;  ws += (size_t)P_PAIRS * sizeof(float);
    float* pL      = (float*)ws;  ws += 256 * sizeof(float);
    float* pM      = (float*)ws;

    convert_w_kernel<<<(D * D) / 256, 256, 0, stream>>>(Ws, Wd, Wsb, Wdb);
    proj_norm_kernel<<<N_NODES / ROWS_PB, 256, 0, stream>>>(emb, Wsb, Wdb, Ub, Vb);
    pair_loss_kernel<<<P_PAIRS / 8, 256, 0, stream>>>(Ub, Vb, pos_src, pos_dst, pos_sign,
                                                      neg_src, neg_dst, lossArr, maskArr);
    reduce_stage1<<<P_PAIRS / 256, 256, 0, stream>>>(lossArr, maskArr, pL, pM);
    reduce_stage2<<<1, 256, 0, stream>>>(pL, pM, (float*)d_out);
}
